// TransformerLayer_29609504539280
// MI455X (gfx1250) — compile-verified, matrix operand order checked by reading, not run on hardware
//
#include <hip/hip_runtime.h>
#include <hip/hip_bf16.h>
#include <math.h>

// ---------------------------------------------------------------------------
// Transformer layer for MI455X (gfx1250, wave32, WMMA).
// All matmuls use v_wmma_f32_16x16x32_bf16 (bf16 in, f32 accum).
// Weights pre-transposed at cast time so GEMM A and B tiles are both
// contiguous-row copies -> double-buffered async global->LDS staging.
// ---------------------------------------------------------------------------

typedef __attribute__((ext_vector_type(16))) __bf16 v16bf;
typedef __attribute__((ext_vector_type(8)))  float  v8f;
typedef __attribute__((ext_vector_type(4)))  float  f32x4;
typedef __attribute__((ext_vector_type(4)))  int    i32x4;

#define DMODEL 1024
#define DHEAD  64
#define NHEADS 16
#define DHID   4096
#define SEQ    2048
#define NTOK   4096   // B * S

union Frag16 { f32x4 f[2]; v16bf v; };
union Half8  { f32x4 f; __bf16 b[8]; };

#if __has_builtin(__builtin_amdgcn_global_load_async_to_lds_b128)
#define HAVE_ASYNC_LDS 1
#else
#define HAVE_ASYNC_LDS 0
#endif

// Copy 16 bytes (8 bf16) global -> LDS; async (per-lane gather) when available.
__device__ __forceinline__ void async_copy16(const __bf16* g, __bf16* l) {
#if HAVE_ASYNC_LDS
  __builtin_amdgcn_global_load_async_to_lds_b128(
      (__attribute__((address_space(1))) i32x4*)g,
      (__attribute__((address_space(3))) i32x4*)l, 0, 0);
#else
  *(f32x4*)l = *(const f32x4*)g;
#endif
}

__device__ __forceinline__ void wait_async_all() {
  asm volatile("s_wait_asynccnt 0x0" ::: "memory");
}

__device__ __forceinline__ v16bf load_frag(const __bf16* p0, const __bf16* p1) {
  Frag16 u;
  u.f[0] = *(const f32x4*)p0;   // elems 0..7  (K = half+0..7)
  u.f[1] = *(const f32x4*)p1;   // elems 8..15 (K = 16+half+0..7)
  return u.v;
}

__device__ __forceinline__ v8f zero8() {
  v8f z = {0.f,0.f,0.f,0.f,0.f,0.f,0.f,0.f};
  return z;
}

__device__ __forceinline__ v8f wmma_bf16(v16bf a, v16bf b, v8f c) {
  return __builtin_amdgcn_wmma_f32_16x16x32_bf16(
      false, a, false, b, (short)0, c, false, false);
}

// ---------------------------------------------------------------------------
// fp32 -> bf16 cast (n divisible by 1024)
// ---------------------------------------------------------------------------
__global__ __launch_bounds__(256)
void cast_bf16_kernel(const float* __restrict__ s, __bf16* __restrict__ d, int n) {
  int i = (blockIdx.x * 256 + threadIdx.x) * 4;
  if (i >= n) return;
  f32x4 v = *(const f32x4*)(s + i);
  d[i + 0] = (__bf16)v.x;
  d[i + 1] = (__bf16)v.y;
  d[i + 2] = (__bf16)v.z;
  d[i + 3] = (__bf16)v.w;
}

// ---------------------------------------------------------------------------
// fp32 W[K,N] -> bf16 Wt[N,K] (transpose via LDS tile). Grid (N/32, K/32).
// ---------------------------------------------------------------------------
__global__ __launch_bounds__(256)
void cast_transpose_kernel(const float* __restrict__ W, __bf16* __restrict__ Wt,
                           int K, int N) {
  __shared__ float tile[32][33];
  const int n0 = blockIdx.x * 32, k0 = blockIdx.y * 32;
  const int tx = threadIdx.x & 31, ty = threadIdx.x >> 5;   // 32 x 8
#pragma unroll
  for (int i = 0; i < 32; i += 8)
    tile[ty + i][tx] = W[(size_t)(k0 + ty + i) * N + n0 + tx];
  __syncthreads();
#pragma unroll
  for (int i = 0; i < 32; i += 8)
    Wt[(size_t)(n0 + ty + i) * K + k0 + tx] = (__bf16)tile[tx][ty + i];
}

// ---------------------------------------------------------------------------
// Tiled bf16 WMMA GEMM: C[M,N] = act(A[M,K] @ Bt[N,K]^T + bias)
// Block tile 128x128, k-step 64, double-buffered async LDS staging.
// 256 threads = 8 waves, wave grid 4x2 (MxN); each wave: 2x4 16x16 tiles.
// Both LDS tiles are [row/col][k] with stride 72 so every fragment is two
// contiguous ds_load_b128 per lane (A layout: K groups of 8 per lane-half).
// ---------------------------------------------------------------------------
template<bool GELU, bool OUT_BF16>
__global__ __launch_bounds__(256)
void gemm_kernel(const __bf16* __restrict__ A, const __bf16* __restrict__ Bt,
                 const float* __restrict__ bias, void* __restrict__ Cout,
                 int M, int N, int K)
{
  __shared__ __bf16 As[2][128 * 72];
  __shared__ __bf16 Bs[2][128 * 72];

  const int tid   = threadIdx.x;
  const int lane  = tid & 31;
  const int wave  = tid >> 5;
  const int waveM = wave >> 1;           // 0..3
  const int waveN = wave & 1;            // 0..1
  const int lr    = lane & 15;
  const int half  = (lane >> 4) * 8;
  const int blkN  = blockIdx.x * 128;
  const int blkM  = blockIdx.y * 128;

  v8f acc[2][4];
#pragma unroll
  for (int tm = 0; tm < 2; ++tm)
#pragma unroll
    for (int tn = 0; tn < 4; ++tn) acc[tm][tn] = zero8();

  const int srow = tid >> 1;             // 0..127
  const int skc  = (tid & 1) * 32;       // 0,32

  auto stage = [&](int k0, int buf) {
    const __bf16* ag = A  + (size_t)(blkM + srow) * K + k0 + skc;
    const __bf16* bg = Bt + (size_t)(blkN + srow) * K + k0 + skc;
    __bf16* al = &As[buf][srow * 72 + skc];
    __bf16* bl = &Bs[buf][srow * 72 + skc];
#pragma unroll
    for (int e = 0; e < 4; ++e) {
      async_copy16(ag + e * 8, al + e * 8);
      async_copy16(bg + e * 8, bl + e * 8);
    }
  };

  stage(0, 0);
  const int T = K >> 6;                  // K / 64
  for (int i = 0; i < T; ++i) {
    wait_async_all();                    // own tile-i loads landed
    __syncthreads();                     // everyone's tile-i landed; prev reads done
    if (i + 1 < T) stage((i + 1) * 64, (i + 1) & 1);

    const __bf16* as = As[i & 1];
    const __bf16* bs = Bs[i & 1];
#pragma unroll
    for (int kc = 0; kc < 2; ++kc) {
      v16bf af[2], bfv[4];
#pragma unroll
      for (int tm = 0; tm < 2; ++tm) {
        const __bf16* p = as + (waveM * 32 + tm * 16 + lr) * 72 + kc * 32 + half;
        af[tm] = load_frag(p, p + 16);
      }
#pragma unroll
      for (int tn = 0; tn < 4; ++tn) {
        const __bf16* p = bs + (waveN * 64 + tn * 16 + lr) * 72 + kc * 32 + half;
        bfv[tn] = load_frag(p, p + 16);
      }
#pragma unroll
      for (int tm = 0; tm < 2; ++tm)
#pragma unroll
        for (int tn = 0; tn < 4; ++tn)
          acc[tm][tn] = wmma_bf16(af[tm], bfv[tn], acc[tm][tn]);
    }
  }

  // epilogue: bias + optional exact GELU, f32 or bf16 store
#pragma unroll
  for (int tm = 0; tm < 2; ++tm) {
#pragma unroll
    for (int tn = 0; tn < 4; ++tn) {
      int col = blkN + waveN * 64 + tn * 16 + lr;
      float bv = bias ? bias[col] : 0.f;
#pragma unroll
      for (int r = 0; r < 8; ++r) {
        int row = blkM + waveM * 32 + tm * 16 + half + r;
        float v = acc[tm][tn][r] + bv;
        if (GELU) v = 0.5f * v * (1.f + erff(v * 0.70710678118654752f));
        if (OUT_BF16) ((__bf16*)Cout)[(size_t)row * N + col] = (__bf16)v;
        else          ((float*)Cout)[(size_t)row * N + col]  = v;
      }
    }
  }
}

// ---------------------------------------------------------------------------
// Causal flash attention. Grid: (SEQ/64, NHEADS, B). 128 threads = 4 waves.
// Each wave owns 16 query rows x 64 d_v. 32-key tiles; K staged async,
// V staged transposed. Q*K^T and P*V via WMMA; online softmax with 16-lane
// shfl_xor row reductions; P routed through LDS into A-fragment layout.
// ---------------------------------------------------------------------------
__global__ __launch_bounds__(128)
void attn_kernel(const __bf16* __restrict__ qb, const __bf16* __restrict__ kb,
                 const __bf16* __restrict__ vb, __bf16* __restrict__ ob)
{
  __shared__ __bf16 Kt[32 * 72];        // [key][d]
  __shared__ __bf16 Vt[64 * 40];        // [d][key] transposed
  __shared__ __bf16 Pb[4 * 16 * 40];    // per-wave P scratch

  const int tid  = threadIdx.x;
  const int lane = tid & 31;
  const int wave = tid >> 5;
  const int lr   = lane & 15;
  const int half = (lane >> 4) * 8;

  const int head = blockIdx.y;
  const int q0   = blockIdx.x * 64;
  const int qw   = q0 + wave * 16;
  const size_t tokbase = (size_t)blockIdx.z * SEQ;

  v16bf qf[2];
#pragma unroll
  for (int di = 0; di < 2; ++di) {
    const __bf16* p = qb + (tokbase + qw + lr) * DMODEL + head * DHEAD + di * 32 + half;
    qf[di] = load_frag(p, p + 16);
  }

  float m[8], l[8];
  v8f o[4];
#pragma unroll
  for (int r = 0; r < 8; ++r) { m[r] = -1e30f; l[r] = 0.f; }
#pragma unroll
  for (int t = 0; t < 4; ++t) o[t] = zero8();

  const int ktmax = (q0 + 63) >> 5;
  for (int kt = 0; kt <= ktmax; ++kt) {
    const int kpos = kt * 32;
#pragma unroll
    for (int c = tid; c < 256; c += 128) {
      int row = c >> 3;                 // key 0..31
      int off = (c & 7) * 8;            // d 0..56
      async_copy16(kb + (tokbase + kpos + row) * DMODEL + head * DHEAD + off,
                   Kt + row * 72 + off);
      Half8 tv;
      tv.f = *(const f32x4*)(vb + (tokbase + kpos + row) * DMODEL + head * DHEAD + off);
#pragma unroll
      for (int e = 0; e < 8; ++e) Vt[(off + e) * 40 + row] = tv.b[e];
    }
    wait_async_all();
    __syncthreads();

    // S = Q * K^T (16x32) as two 16x16 tiles
    v8f s0 = zero8(), s1 = zero8();
#pragma unroll
    for (int di = 0; di < 2; ++di) {
      const __bf16* p0 = Kt + lr * 72        + di * 32 + half;
      const __bf16* p1 = Kt + (16 + lr) * 72 + di * 32 + half;
      v16bf b0 = load_frag(p0, p0 + 16);
      v16bf b1 = load_frag(p1, p1 + 16);
      s0 = wmma_bf16(qf[di], b0, s0);
      s1 = wmma_bf16(qf[di], b1, s1);
    }

    // online softmax
    float p0v[8], p1v[8], alpha[8];
#pragma unroll
    for (int r = 0; r < 8; ++r) {
      int qrow = qw + half + r;
      int c0 = kpos + lr, c1 = kpos + 16 + lr;
      bool ok0 = (c0 <= qrow), ok1 = (c1 <= qrow);
      float v0 = ok0 ? s0[r] * 0.125f : -1e30f;
      float v1 = ok1 ? s1[r] * 0.125f : -1e30f;
      float mx = fmaxf(v0, v1);
#pragma unroll
      for (int msk = 8; msk >= 1; msk >>= 1) mx = fmaxf(mx, __shfl_xor(mx, msk, 32));
      float mn = fmaxf(m[r], mx);
      alpha[r] = expf(m[r] - mn);
      float p0 = ok0 ? expf(v0 - mn) : 0.f;
      float p1 = ok1 ? expf(v1 - mn) : 0.f;
      float ps = p0 + p1;
#pragma unroll
      for (int msk = 8; msk >= 1; msk >>= 1) ps += __shfl_xor(ps, msk, 32);
      l[r] = l[r] * alpha[r] + ps;
      m[r] = mn;
      p0v[r] = p0; p1v[r] = p1;
    }
#pragma unroll
    for (int t = 0; t < 4; ++t)
#pragma unroll
      for (int r = 0; r < 8; ++r) o[t][r] *= alpha[r];

    // P -> LDS -> A-fragment
    __bf16* pw = Pb + wave * (16 * 40);
#pragma unroll
    for (int r = 0; r < 8; ++r) {
      pw[(half + r) * 40 + lr]      = (__bf16)p0v[r];
      pw[(half + r) * 40 + 16 + lr] = (__bf16)p1v[r];
    }
    asm volatile("s_wait_dscnt 0" ::: "memory");

    const __bf16* pa = pw + lr * 40 + half;
    v16bf af = load_frag(pa, pa + 16);
#pragma unroll
    for (int t = 0; t < 4; ++t) {
      const __bf16* pv = Vt + (t * 16 + lr) * 40 + half;
      v16bf bv = load_frag(pv, pv + 16);
      o[t] = wmma_bf16(af, bv, o[t]);
    }
    __syncthreads();
  }

#pragma unroll
  for (int t = 0; t < 4; ++t)
#pragma unroll
    for (int r = 0; r < 8; ++r) {
      float val = o[t][r] / l[r];
      ob[(tokbase + qw + half + r) * DMODEL + head * DHEAD + t * 16 + lr] = (__bf16)val;
    }
}

// ---------------------------------------------------------------------------
// out = layernorm(x + y); optional bf16 copy for the next GEMM.
// ---------------------------------------------------------------------------
__global__ __launch_bounds__(256)
void add_ln_kernel(const float* __restrict__ x, const float* __restrict__ y,
                   const float* __restrict__ g, const float* __restrict__ b,
                   float* __restrict__ out, __bf16* __restrict__ outb)
{
  __shared__ float red[256];
  const int tid = threadIdx.x;
  const size_t base = (size_t)blockIdx.x * DMODEL + tid * 4;

  f32x4 xv = *(const f32x4*)(x + base);
  f32x4 yv = *(const f32x4*)(y + base);
  float v[4];
  float s = 0.f;
#pragma unroll
  for (int i = 0; i < 4; ++i) { v[i] = xv[i] + yv[i]; s += v[i]; }

  red[tid] = s; __syncthreads();
  for (int o_ = 128; o_ > 0; o_ >>= 1) { if (tid < o_) red[tid] += red[tid + o_]; __syncthreads(); }
  float mu = red[0] * (1.f / DMODEL);
  __syncthreads();

  s = 0.f;
#pragma unroll
  for (int i = 0; i < 4; ++i) { float d = v[i] - mu; s += d * d; }
  red[tid] = s; __syncthreads();
  for (int o_ = 128; o_ > 0; o_ >>= 1) { if (tid < o_) red[tid] += red[tid + o_]; __syncthreads(); }
  float rstd = rsqrtf(red[0] * (1.f / DMODEL) + 1e-5f);

  const int col = tid * 4;
#pragma unroll
  for (int i = 0; i < 4; ++i) {
    float o_ = (v[i] - mu) * rstd * g[col + i] + b[col + i];
    out[base + i] = o_;
    if (outb) outb[base + i] = (__bf16)o_;
  }
}

// ---------------------------------------------------------------------------
// Launcher
// ---------------------------------------------------------------------------
extern "C" void kernel_launch(void* const* d_in, const int* in_sizes, int n_in,
                              void* d_out, int out_size, void* d_ws, size_t ws_size,
                              hipStream_t stream) {
  const float* x    = (const float*)d_in[0];
  const float* Wq   = (const float*)d_in[1];
  const float* bq   = (const float*)d_in[2];
  const float* Wk   = (const float*)d_in[3];
  const float* bk   = (const float*)d_in[4];
  const float* Wv   = (const float*)d_in[5];
  const float* bv   = (const float*)d_in[6];
  const float* Wo   = (const float*)d_in[7];
  const float* ln1g = (const float*)d_in[8];
  const float* ln1b = (const float*)d_in[9];
  const float* W1   = (const float*)d_in[10];
  const float* b1   = (const float*)d_in[11];
  const float* W2   = (const float*)d_in[12];
  const float* b2   = (const float*)d_in[13];
  const float* ln2g = (const float*)d_in[14];
  const float* ln2b = (const float*)d_in[15];

  char* ws = (char*)d_ws;
  size_t off = 0;
  auto alloc = [&](size_t bytes) -> char* {
    char* p = ws + off;
    off = (off + bytes + 255) & ~(size_t)255;
    return p;
  };

  __bf16* xb   = (__bf16*)alloc((size_t)NTOK * DMODEL * 2);
  __bf16* Wqt  = (__bf16*)alloc((size_t)DMODEL * DMODEL * 2);   // [N][K]
  __bf16* Wkt  = (__bf16*)alloc((size_t)DMODEL * DMODEL * 2);
  __bf16* Wvt  = (__bf16*)alloc((size_t)DMODEL * DMODEL * 2);
  __bf16* Wot  = (__bf16*)alloc((size_t)DMODEL * DMODEL * 2);
  __bf16* W1t  = (__bf16*)alloc((size_t)DMODEL * DHID * 2);
  __bf16* W2t  = (__bf16*)alloc((size_t)DHID * DMODEL * 2);
  __bf16* qbuf = (__bf16*)alloc((size_t)NTOK * DMODEL * 2);
  __bf16* kbuf = (__bf16*)alloc((size_t)NTOK * DMODEL * 2);
  __bf16* vbuf = (__bf16*)alloc((size_t)NTOK * DMODEL * 2);
  __bf16* obuf = (__bf16*)alloc((size_t)NTOK * DMODEL * 2);
  float*  t1   = (float*)alloc((size_t)NTOK * DMODEL * 4);
  float*  h    = (float*)alloc((size_t)NTOK * DMODEL * 4);
  __bf16* hb   = (__bf16*)alloc((size_t)NTOK * DMODEL * 2);
  __bf16* m1   = (__bf16*)alloc((size_t)NTOK * DHID * 2);
  float*  m2   = (float*)alloc((size_t)NTOK * DMODEL * 4);

  cast_bf16_kernel<<<NTOK * DMODEL / 1024, 256, 0, stream>>>(x, xb, NTOK * DMODEL);

  auto castT = [&](const float* W, __bf16* Wt, int K, int N) {
    cast_transpose_kernel<<<dim3(N / 32, K / 32), 256, 0, stream>>>(W, Wt, K, N);
  };
  castT(Wq, Wqt, DMODEL, DMODEL);
  castT(Wk, Wkt, DMODEL, DMODEL);
  castT(Wv, Wvt, DMODEL, DMODEL);
  castT(Wo, Wot, DMODEL, DMODEL);
  castT(W1, W1t, DMODEL, DHID);
  castT(W2, W2t, DHID, DMODEL);

  // QKV projections (bf16 out)
  gemm_kernel<false, true><<<dim3(DMODEL / 128, NTOK / 128), 256, 0, stream>>>(
      xb, Wqt, bq, qbuf, NTOK, DMODEL, DMODEL);
  gemm_kernel<false, true><<<dim3(DMODEL / 128, NTOK / 128), 256, 0, stream>>>(
      xb, Wkt, bk, kbuf, NTOK, DMODEL, DMODEL);
  gemm_kernel<false, true><<<dim3(DMODEL / 128, NTOK / 128), 256, 0, stream>>>(
      xb, Wvt, bv, vbuf, NTOK, DMODEL, DMODEL);

  // causal flash attention
  attn_kernel<<<dim3(SEQ / 64, NHEADS, 2), 128, 0, stream>>>(qbuf, kbuf, vbuf, obuf);

  // output projection (f32 out, no bias)
  gemm_kernel<false, false><<<dim3(DMODEL / 128, NTOK / 128), 256, 0, stream>>>(
      obuf, Wot, nullptr, t1, NTOK, DMODEL, DMODEL);

  // h = LN(x + attn)
  add_ln_kernel<<<NTOK, 256, 0, stream>>>(x, t1, ln1g, ln1b, h, hb);

  // MLP up with fused exact GELU (bf16 out)
  gemm_kernel<true, true><<<dim3(DHID / 128, NTOK / 128), 256, 0, stream>>>(
      hb, W1t, b1, m1, NTOK, DHID, DMODEL);

  // MLP down (f32 out)
  gemm_kernel<false, false><<<dim3(DMODEL / 128, NTOK / 128), 256, 0, stream>>>(
      m1, W2t, b2, m2, NTOK, DMODEL, DHID);

  // out = LN(h + mlp)
  add_ln_kernel<<<NTOK, 256, 0, stream>>>(h, m2, ln2g, ln2b, (float*)d_out, nullptr);
}
